// Seq2Seq_72559177499106
// MI455X (gfx1250) — matrix operand order repaired and checked by reading, not compiled, not run on hardware
//
#include <hip/hip_runtime.h>

typedef __bf16 bf16_t;
typedef __attribute__((ext_vector_type(16))) __bf16 v16bf;
typedef __attribute__((ext_vector_type(8)))  __bf16 v8bf;
typedef __attribute__((ext_vector_type(8)))  float  v8f;

#define HDIM   512
#define FOURH  2048
#define BATCH  64
#define DMODEL 176
#define DPAD   192            // DMODEL padded to a multiple of 32 (WMMA K)
#define TLEN   1024
#define ENC_STEPS 1025        // x_pad has 1025 rows
#define DEC_STEPS 1023        // y_pad[:, :T-1] has 1023 rows
#define MROWS  (DEC_STEPS * BATCH)   // 65472 classifier rows

// ---------------------------------------------------------------- fragments
// All K extents are multiples of 32 (padded), so loads are unconditional,
// aligned vector loads: A = two 16B loads, B = one 32B load.

__device__ __forceinline__ v16bf load_a_frag(const bf16_t* xrow, int kb, int half) {
  // A 16x32 bf16: lane holds K = kb + half*16 + 0..15 (contiguous)
  v8bf lo = *(const v8bf*)(xrow + kb + half * 16);
  v8bf hi = *(const v8bf*)(xrow + kb + half * 16 + 8);
  v16bf a;
#pragma unroll
  for (int i = 0; i < 8; ++i) { a[i] = lo[i]; a[8 + i] = hi[i]; }
  return a;
}

__device__ __forceinline__ v16bf load_b_frag(const bf16_t* wrow, int kb, int half) {
  // B 32x16 bf16: lane holds K = kb + half*16 + 0..15 (contiguous)
  return *(const v16bf*)(wrow + kb + half * 16);
}

// ------------------------------------------------------- fused LSTM step

struct StepLayer {
  const bf16_t* x;      // layer input (bf16); row b at x + b*xStride; null => zeros
  long          xStride;
  int           Din;    // K extent for x-part (multiple of 32)
  const bf16_t* Wih;    // [4H][Din] bf16
  const bf16_t* Whh;    // [4H][H]   bf16
  const float*  bih;    // [4H] f32
  const float*  bhh;    // [4H] f32
  const bf16_t* hprev;  // [B][H] bf16; null => zeros
  float*        c;      // [B][H] f32 in-out (carry)
  int           c_init; // 1 => c_old = 0
  bf16_t*       hout;   // [B][H] bf16
  bf16_t*       hout2;  // optional extra store (dec_out slice) or null
};

__device__ __forceinline__ void accum_gates(v8f acc[4],
                                            const bf16_t* xbase, long xStride,
                                            const bf16_t* W, int Din,
                                            int row0, int htile, int lane) {
  int ncol = lane & 15, half = lane >> 4;
  const bf16_t* xrow = xbase + (long)(row0 + ncol) * xStride;  // A row m = row0 + lane%16
  for (int kb = 0; kb < Din; kb += 32) {
    v16bf a = load_a_frag(xrow, kb, half);
#pragma unroll
    for (int g = 0; g < 4; ++g) {
      const bf16_t* wrow = W + (long)(g * HDIM + htile * 16 + ncol) * Din;
      v16bf b = load_b_frag(wrow, kb, half);
      acc[g] = __builtin_amdgcn_wmma_f32_16x16x32_bf16(
          false, a, false, b, (short)0, acc[g], false, false);
    }
  }
}

// One launch = 32 blocks x 256 thr. Blocks 0..15: layer0 @ step t; 16..31: layer1 @ step t-1.
// 128 waves per layer = 4 batch-tiles x 32 h-tiles; each wave computes the i/f/g/o
// 16x16 tiles for its (batch,h) range with 4 WMMAs per K-chunk (shared A fragment),
// then does the gate nonlinearity + c/h update fully fused in registers.
__global__ __launch_bounds__(256) void lstm_step_kernel(StepLayer L0, StepLayer L1,
                                                        int act0, int act1) {
  int layer = blockIdx.x >> 4;
  StepLayer L = layer ? L1 : L0;
  if (!(layer ? act1 : act0)) return;

  int blk  = blockIdx.x & 15;
  int tid  = threadIdx.x;
  int lane = tid & 31, widx = tid >> 5;
  int wg    = blk * 8 + widx;     // 0..127
  int mtile = wg & 3;             // 64 batch rows / 16
  int htile = wg >> 2;            // 512 h cols / 16
  int row0  = mtile * 16;
  int ncol  = lane & 15, half = lane >> 4;

  v8f acc[4] = {};

  if (L.x)     accum_gates(acc, L.x,     L.xStride, L.Wih, L.Din, row0, htile, lane);
  if (L.hprev) accum_gates(acc, L.hprev, HDIM,      L.Whh, HDIM,  row0, htile, lane);

  int j = htile * 16 + ncol;
  float bI = L.bih[j]            + L.bhh[j];
  float bF = L.bih[HDIM + j]     + L.bhh[HDIM + j];
  float bG = L.bih[2 * HDIM + j] + L.bhh[2 * HDIM + j];
  float bO = L.bih[3 * HDIM + j] + L.bhh[3 * HDIM + j];

  float hvals[8];
#pragma unroll
  for (int v = 0; v < 8; ++v) {
    int  m   = row0 + v + half * 8;           // C/D layout: M = v + 8*(lane/16)
    long idx = (long)m * HDIM + j;
    float gi = acc[0][v] + bI;
    float gf = acc[1][v] + bF;
    float gg = acc[2][v] + bG;
    float go = acc[3][v] + bO;
    float cold = L.c_init ? 0.0f : L.c[idx];
    float si = 1.0f / (1.0f + __expf(-gi));
    float sf = 1.0f / (1.0f + __expf(-gf));
    float so = 1.0f / (1.0f + __expf(-go));
    float cn = sf * cold + si * tanhf(gg);
    float hn = so * tanhf(cn);
    L.c[idx]    = cn;
    L.hout[idx] = (bf16_t)hn;
    hvals[v]    = hn;
  }
  if (L.hout2) {
#pragma unroll
    for (int v = 0; v < 8; ++v) {
      int m = row0 + v + half * 8;
      L.hout2[(long)m * HDIM + j] = (bf16_t)hvals[v];
    }
  }
}

// ------------------------------------------------------- classifier GEMMs

// hmid = relu(dec_out @ W1^T + b1) : M=65472, N=256, K=512
__global__ __launch_bounds__(256) void cls1_kernel(const bf16_t* A, const bf16_t* W,
                                                   const float* b1, bf16_t* hmid) {
  int tid = threadIdx.x, lane = tid & 31, widx = tid >> 5;
  int wg = blockIdx.x * 8 + widx;
  int ntile = wg & 15;           // 256/16
  int mtile = wg >> 4;
  if (mtile * 16 >= MROWS) return;
  int ncol = lane & 15, half = lane >> 4;
  const bf16_t* arow = A + (long)(mtile * 16 + ncol) * HDIM;
  const bf16_t* wrow = W + (long)(ntile * 16 + ncol) * HDIM;
  v8f acc = {};
  for (int kb = 0; kb < HDIM; kb += 32) {
    v16bf a = load_a_frag(arow, kb, half);
    v16bf b = load_b_frag(wrow, kb, half);
    acc = __builtin_amdgcn_wmma_f32_16x16x32_bf16(false, a, false, b, (short)0, acc,
                                                  false, false);
  }
  int   col = ntile * 16 + ncol;
  float bb  = b1[col];
#pragma unroll
  for (int v = 0; v < 8; ++v) {
    int   r   = mtile * 16 + v + half * 8;
    float val = acc[v] + bb;
    hmid[(long)r * 256 + col] = (bf16_t)(val > 0.0f ? val : 0.0f);
  }
}

// logits = hmid @ W2^T + b2, scattered into out[b, t+1, :] : M=65472, N=176, K=256
__global__ __launch_bounds__(256) void cls2_kernel(const bf16_t* A, const bf16_t* W,
                                                   const float* b2, float* out) {
  int tid = threadIdx.x, lane = tid & 31, widx = tid >> 5;
  int wg = blockIdx.x * 8 + widx;
  const int NT = 11;             // ceil(176/16)
  int ntile = wg % NT;
  int mtile = wg / NT;
  if (mtile * 16 >= MROWS) return;
  int ncol = lane & 15, half = lane >> 4;
  int d = ntile * 16 + ncol;
  const bf16_t* arow = A + (long)(mtile * 16 + ncol) * 256;
  const bf16_t* wrow = (d < DMODEL) ? (W + (long)d * 256) : W;  // dummy row for OOB cols
  v8f acc = {};
  for (int kb = 0; kb < 256; kb += 32) {
    v16bf a = load_a_frag(arow, kb, half);
    v16bf b = load_b_frag(wrow, kb, half);
    acc = __builtin_amdgcn_wmma_f32_16x16x32_bf16(false, a, false, b, (short)0, acc,
                                                  false, false);
  }
  if (d < DMODEL) {
    float bb = b2[d];
#pragma unroll
    for (int v = 0; v < 8; ++v) {
      int r = mtile * 16 + v + half * 8;   // r = t*64 + b
      int t = r >> 6, b = r & 63;
      out[((long)b * TLEN + (t + 1)) * DMODEL + d] = acc[v] + bb;
    }
  }
}

// ------------------------------------------------------- small utilities

// Plain fp32 -> bf16 (same shape)
__global__ void cvt_bf16_kernel(const float* src, bf16_t* dst, long n) {
  long i = (long)blockIdx.x * blockDim.x + threadIdx.x;
  if (i < n) dst[i] = (bf16_t)src[i];
}

// Row-wise fp32 -> bf16 with zero-padded columns (srcCols -> dstCols)
__global__ void cvt_pad_bf16_kernel(const float* src, bf16_t* dst, long rows,
                                    int srcCols, int dstCols) {
  long i = (long)blockIdx.x * blockDim.x + threadIdx.x;
  if (i < rows * (long)dstCols) {
    long r = i / dstCols;
    int  k = (int)(i - r * dstCols);
    dst[i] = (k < srcCols) ? (bf16_t)src[r * srcCols + k] : (bf16_t)0.0f;
  }
}

__global__ void zero_row0_kernel(float* out) {
  int i = blockIdx.x * blockDim.x + threadIdx.x;
  if (i < BATCH * DMODEL) {
    int b = i / DMODEL, d = i % DMODEL;
    out[(long)b * TLEN * DMODEL + d] = 0.0f;
  }
}

// ------------------------------------------------------- host driver

extern "C" void kernel_launch(void* const* d_in, const int* in_sizes, int n_in,
                              void* d_out, int out_size, void* d_ws, size_t ws_size,
                              hipStream_t stream) {
  (void)in_sizes; (void)n_in; (void)out_size; (void)ws_size;
  const float* f_x   = (const float*)d_in[0];
  const float* f_y   = (const float*)d_in[1];
  const float* eWih0 = (const float*)d_in[2];
  const float* eWhh0 = (const float*)d_in[3];
  const float* ebih0 = (const float*)d_in[4];
  const float* ebhh0 = (const float*)d_in[5];
  const float* eWih1 = (const float*)d_in[6];
  const float* eWhh1 = (const float*)d_in[7];
  const float* ebih1 = (const float*)d_in[8];
  const float* ebhh1 = (const float*)d_in[9];
  const float* dWih0 = (const float*)d_in[10];
  const float* dWhh0 = (const float*)d_in[11];
  const float* dbih0 = (const float*)d_in[12];
  const float* dbhh0 = (const float*)d_in[13];
  const float* dWih1 = (const float*)d_in[14];
  const float* dWhh1 = (const float*)d_in[15];
  const float* dbih1 = (const float*)d_in[16];
  const float* dbhh1 = (const float*)d_in[17];
  const float* cW1   = (const float*)d_in[18];
  const float* cb1   = (const float*)d_in[19];
  const float* cW2   = (const float*)d_in[20];
  const float* cb2   = (const float*)d_in[21];
  float* out = (float*)d_out;

  char*  ws  = (char*)d_ws;
  size_t off = 0;
  auto carve = [&](size_t bytes) -> char* {
    char* p = ws + off;
    off = (off + bytes + 255) & ~(size_t)255;
    return p;
  };

  // bf16 weight / activation copies (K-padded where K=176)
  bf16_t* b_eWih0 = (bf16_t*)carve((size_t)FOURH * DPAD * 2);
  bf16_t* b_eWhh0 = (bf16_t*)carve((size_t)FOURH * HDIM * 2);
  bf16_t* b_eWih1 = (bf16_t*)carve((size_t)FOURH * HDIM * 2);
  bf16_t* b_eWhh1 = (bf16_t*)carve((size_t)FOURH * HDIM * 2);
  bf16_t* b_dWih0 = (bf16_t*)carve((size_t)FOURH * DPAD * 2);
  bf16_t* b_dWhh0 = (bf16_t*)carve((size_t)FOURH * HDIM * 2);
  bf16_t* b_dWih1 = (bf16_t*)carve((size_t)FOURH * HDIM * 2);
  bf16_t* b_dWhh1 = (bf16_t*)carve((size_t)FOURH * HDIM * 2);
  bf16_t* b_cW1   = (bf16_t*)carve((size_t)256 * HDIM * 2);
  bf16_t* b_cW2   = (bf16_t*)carve((size_t)DMODEL * 256 * 2);
  bf16_t* b_x     = (bf16_t*)carve((size_t)BATCH * TLEN * DPAD * 2);
  bf16_t* b_y     = (bf16_t*)carve((size_t)BATCH * TLEN * DPAD * 2);

  // state + intermediate buffers
  bf16_t* h0buf[2] = {(bf16_t*)carve(BATCH * HDIM * 2), (bf16_t*)carve(BATCH * HDIM * 2)};
  bf16_t* h1buf[2] = {(bf16_t*)carve(BATCH * HDIM * 2), (bf16_t*)carve(BATCH * HDIM * 2)};
  bf16_t* hinit0   = (bf16_t*)carve(BATCH * HDIM * 2);
  bf16_t* hinit1   = (bf16_t*)carve(BATCH * HDIM * 2);
  float*  c0       = (float*)carve(BATCH * HDIM * 4);
  float*  c1       = (float*)carve(BATCH * HDIM * 4);
  bf16_t* dec_out  = (bf16_t*)carve((size_t)MROWS * HDIM * 2);
  bf16_t* hmid     = (bf16_t*)carve((size_t)MROWS * 256 * 2);

  // fp32 -> bf16 conversion pass (padded variants for K=176 operands)
  struct PadJob { const float* s; bf16_t* d; long rows; int sc; int dc; };
  PadJob pjobs[] = {
      {eWih0, b_eWih0, FOURH, DMODEL, DPAD},
      {dWih0, b_dWih0, FOURH, DMODEL, DPAD},
      {f_x,   b_x,     (long)BATCH * TLEN, DMODEL, DPAD},
      {f_y,   b_y,     (long)BATCH * TLEN, DMODEL, DPAD},
  };
  for (auto& j : pjobs) {
    long n = j.rows * (long)j.dc;
    cvt_pad_bf16_kernel<<<(int)((n + 255) / 256), 256, 0, stream>>>(j.s, j.d, j.rows,
                                                                    j.sc, j.dc);
  }
  struct CvtJob { const float* s; bf16_t* d; long n; };
  CvtJob jobs[] = {
      {eWhh0, b_eWhh0, (long)FOURH * HDIM}, {eWih1, b_eWih1, (long)FOURH * HDIM},
      {eWhh1, b_eWhh1, (long)FOURH * HDIM}, {dWhh0, b_dWhh0, (long)FOURH * HDIM},
      {dWih1, b_dWih1, (long)FOURH * HDIM}, {dWhh1, b_dWhh1, (long)FOURH * HDIM},
      {cW1,   b_cW1,   (long)256 * HDIM},   {cW2,   b_cW2,   (long)DMODEL * 256},
  };
  for (auto& j : jobs) {
    cvt_bf16_kernel<<<(int)((j.n + 255) / 256), 256, 0, stream>>>(j.s, j.d, j.n);
  }

  // ---- encoder: software-pipelined layers, 1 launch per step --------------
  for (int s = 0; s <= ENC_STEPS; ++s) {
    int act0 = (s < ENC_STEPS) ? 1 : 0;  // layer0 at t=s
    int act1 = (s >= 1) ? 1 : 0;         // layer1 at t=s-1
    StepLayer L0 = {}, L1 = {};
    if (act0) {
      int t = s;
      L0.x       = (t == 0) ? nullptr : b_x + (long)(t - 1) * DPAD;
      L0.xStride = (long)TLEN * DPAD;
      L0.Din     = DPAD;
      L0.Wih = b_eWih0; L0.Whh = b_eWhh0; L0.bih = ebih0; L0.bhh = ebhh0;
      L0.hprev  = (t == 0) ? nullptr : h0buf[(t - 1) & 1];
      L0.c      = c0;  L0.c_init = (t == 0);
      L0.hout   = h0buf[t & 1]; L0.hout2 = nullptr;
    }
    if (act1) {
      int t = s - 1;
      L1.x       = h0buf[t & 1];
      L1.xStride = HDIM;
      L1.Din     = HDIM;
      L1.Wih = b_eWih1; L1.Whh = b_eWhh1; L1.bih = ebih1; L1.bhh = ebhh1;
      L1.hprev  = (t == 0) ? nullptr : h1buf[(t - 1) & 1];
      L1.c      = c1;  L1.c_init = (t == 0);
      L1.hout   = h1buf[t & 1]; L1.hout2 = nullptr;
    }
    lstm_step_kernel<<<32, 256, 0, stream>>>(L0, L1, act0, act1);
  }

  // snapshot encoder-final h for decoder init (c0/c1 carry over in place)
  int pf = (ENC_STEPS - 1) & 1;  // parity of final encoder step (t=1024 -> 0)
  hipMemcpyAsync(hinit0, h0buf[pf], BATCH * HDIM * 2, hipMemcpyDeviceToDevice, stream);
  hipMemcpyAsync(hinit1, h1buf[pf], BATCH * HDIM * 2, hipMemcpyDeviceToDevice, stream);

  // ---- decoder ------------------------------------------------------------
  for (int s = 0; s <= DEC_STEPS; ++s) {
    int act0 = (s < DEC_STEPS) ? 1 : 0;
    int act1 = (s >= 1) ? 1 : 0;
    StepLayer L0 = {}, L1 = {};
    if (act0) {
      int t = s;
      L0.x       = (t == 0) ? nullptr : b_y + (long)(t - 1) * DPAD;
      L0.xStride = (long)TLEN * DPAD;
      L0.Din     = DPAD;
      L0.Wih = b_dWih0; L0.Whh = b_dWhh0; L0.bih = dbih0; L0.bhh = dbhh0;
      L0.hprev  = (t == 0) ? hinit0 : h0buf[(t - 1) & 1];
      L0.c      = c0;  L0.c_init = 0;            // carry encoder cT
      L0.hout   = h0buf[t & 1]; L0.hout2 = nullptr;
    }
    if (act1) {
      int t = s - 1;
      L1.x       = h0buf[t & 1];
      L1.xStride = HDIM;
      L1.Din     = HDIM;
      L1.Wih = b_dWih1; L1.Whh = b_dWhh1; L1.bih = dbih1; L1.bhh = dbhh1;
      L1.hprev  = (t == 0) ? hinit1 : h1buf[(t - 1) & 1];
      L1.c      = c1;  L1.c_init = 0;
      L1.hout   = h1buf[t & 1];
      L1.hout2  = dec_out + (long)t * BATCH * HDIM;  // time-major [t][b][h]
    }
    lstm_step_kernel<<<32, 256, 0, stream>>>(L0, L1, act0, act1);
  }

  // ---- classifier: two batched WMMA GEMMs + zero first output row ---------
  {
    int waves1  = (MROWS / 16) * 16;                 // 4092 Mtiles x 16 Ntiles
    int blocks1 = waves1 / 8;
    cls1_kernel<<<blocks1, 256, 0, stream>>>(dec_out, b_cW1, cb1, hmid);

    int waves2  = (MROWS / 16) * 11;                 // 4092 x 11
    int blocks2 = (waves2 + 7) / 8;
    cls2_kernel<<<blocks2, 256, 0, stream>>>(hmid, b_cW2, cb2, out);

    zero_row0_kernel<<<(BATCH * DMODEL + 255) / 256, 256, 0, stream>>>(out);
  }
}
